// Block_7086696039160
// MI455X (gfx1250) — matrix operand order repaired and checked by reading, not compiled, not run on hardware
//
#include <hip/hip_runtime.h>
#include <hip/hip_bf16.h>
#include <math.h>
#include <stdint.h>

// ---------------------------------------------------------------------------
// MoE transformer block for MI455X (gfx1250, wave32, WMMA).
// GEMMs: v_wmma_f32_16x16x32_bf16 with double-buffered
// global_load_async_to_lds_b128 staging (ASYNCcnt) per CDNA5 ISA.
// ---------------------------------------------------------------------------

typedef __attribute__((ext_vector_type(16))) __bf16 bf16x16;
typedef __attribute__((ext_vector_type(8)))  float  f32x8;

union FragBF { uint4 q[2]; bf16x16 v; };

__device__ __forceinline__ f32x8 zero8() {
  f32x8 z = {0.f, 0.f, 0.f, 0.f, 0.f, 0.f, 0.f, 0.f};
  return z;
}

// Async-copy one 128x32 bf16 tile (row stride ld elems) from global into LDS
// with 40-elem padded rows. 256 threads; each thread issues two 16B copies,
// i.e. 4 async instructions per wave per (A,B) tile pair.
__device__ __forceinline__ void async_tile_128x32(const __bf16* g_base,
                                                  unsigned lds_base_bytes,
                                                  int ld, int tid) {
  #pragma unroll
  for (int s = 0; s < 2; ++s) {
    const int seg = tid * 2 + s;          // 0..511
    const int r = seg >> 2;               // row 0..127
    const int c = (seg & 3) * 8;          // elem col 0,8,16,24
    const unsigned long long g =
        (unsigned long long)(uintptr_t)(g_base + (long long)r * ld + c);
    const unsigned l = lds_base_bytes + (unsigned)(r * 40 + c) * 2u;
    asm volatile("global_load_async_to_lds_b128 %0, %1, off"
                 :: "v"(l), "v"(g) : "memory");
  }
}

__device__ __forceinline__ void wait_async_le4() {
  asm volatile("s_wait_asynccnt 0x4" ::: "memory");
}
__device__ __forceinline__ void wait_async_0() {
  asm volatile("s_wait_asynccnt 0x0" ::: "memory");
}

// ---------------------------------------------------------------------------
// Transposing f32 -> bf16 weight conversion: dst[n*K + k] = bf16(src[k*N + n])
// Tiled 32x32 through LDS, coalesced on both sides. grid.z = batch.
// ---------------------------------------------------------------------------
__global__ __launch_bounds__(256)
void k_cvt_t(const float* __restrict__ src, __bf16* __restrict__ dst,
             int K, int N) {
  __shared__ float tile[32][33];
  const long long bofs = (long long)blockIdx.z * K * N;
  src += bofs;
  dst += bofs;
  const int kb = blockIdx.y * 32, nb = blockIdx.x * 32;
  const int tx = threadIdx.x & 31, ty = threadIdx.x >> 5;  // 32 x 8
  #pragma unroll
  for (int i = 0; i < 32; i += 8)
    tile[ty + i][tx] = src[(long long)(kb + ty + i) * N + nb + tx];
  __syncthreads();
  #pragma unroll
  for (int i = 0; i < 32; i += 8)
    dst[(long long)(nb + ty + i) * K + kb + tx] = (__bf16)tile[tx][ty + i];
}

// ---------------------------------------------------------------------------
// LayerNorm over rows of 1024, optional residual add before LN.
// ---------------------------------------------------------------------------
__global__ __launch_bounds__(256)
void k_ln_bf16(const float* __restrict__ x, const float* __restrict__ add,
               const float* __restrict__ gamma, const float* __restrict__ beta,
               float* __restrict__ x1f, float* __restrict__ hf,
               __bf16* __restrict__ hb) {
  __shared__ float2 red[256];
  const int r = blockIdx.x, tid = threadIdx.x;
  float4 v = ((const float4*)(x + (long long)r * 1024))[tid];
  if (add) {
    float4 a = ((const float4*)(add + (long long)r * 1024))[tid];
    v.x += a.x; v.y += a.y; v.z += a.z; v.w += a.w;
  }
  if (x1f) ((float4*)(x1f + (long long)r * 1024))[tid] = v;
  float s = v.x + v.y + v.z + v.w;
  float q = v.x * v.x + v.y * v.y + v.z * v.z + v.w * v.w;
  red[tid] = make_float2(s, q);
  __syncthreads();
  for (int off = 128; off > 0; off >>= 1) {
    if (tid < off) {
      red[tid].x += red[tid + off].x;
      red[tid].y += red[tid + off].y;
    }
    __syncthreads();
  }
  const float mean = red[0].x * (1.f / 1024.f);
  const float var = red[0].y * (1.f / 1024.f) - mean * mean;
  const float rstd = rsqrtf(var + 1e-5f);
  const int c = tid * 4;
  float o[4] = {v.x, v.y, v.z, v.w};
  #pragma unroll
  for (int i = 0; i < 4; ++i)
    o[i] = (o[i] - mean) * rstd * gamma[c + i] + beta[c + i];
  if (hf) ((float4*)(hf + (long long)r * 1024))[tid] = make_float4(o[0], o[1], o[2], o[3]);
  __bf16* hbr = hb + (long long)r * 1024 + c;
  #pragma unroll
  for (int i = 0; i < 4; ++i) hbr[i] = (__bf16)o[i];
}

// ---------------------------------------------------------------------------
// Batched bf16 WMMA GEMM: C[M,N] = A[M,K] @ BT[N,K]^T (+bias) (+relu).
// B is supplied PRE-TRANSPOSED so both tiles are contiguous-row copies done
// with double-buffered global_load_async_to_lds_b128 (no VGPR staging, no
// LDS transpose). Block tile 128x128, BK=32; 8 waves x (32x64), 8 WMMAs/step.
// LDS rows padded to 40 elems (80B, 16B-aligned): conflict-free ds_load_b128
// fragment reads per CDNA5 ISA 7.12.2 layouts.
// ---------------------------------------------------------------------------
template <bool HAS_BIAS, bool RELU, bool OUT_F, bool OUT_B>
__global__ __launch_bounds__(256)
void k_gemm_bf16(const __bf16* __restrict__ A, const __bf16* __restrict__ BT,
                 const float* __restrict__ bias,
                 float* __restrict__ Cf, __bf16* __restrict__ Cb,
                 int M, int N, int K,
                 long long sA, long long sB, long long sBias,
                 long long sCf, long long sCb) {
  __shared__ __align__(16) __bf16 As[2][128 * 40];
  __shared__ __align__(16) __bf16 Bs[2][128 * 40];
  const int batch = blockIdx.z;
  const int m0 = blockIdx.y * 128, n0 = blockIdx.x * 128;
  const int tid = threadIdx.x;
  const int wave = tid >> 5, lane = tid & 31;
  const int wm = (wave & 3) * 32, wn = (wave >> 2) * 64;
  const int lh = lane & 15, kh = lane >> 4;

  const __bf16* Ab = A + (long long)batch * sA + (long long)m0 * K;
  const __bf16* Bb = BT + (long long)batch * sB + (long long)n0 * K;
  if (HAS_BIAS) bias += (long long)batch * sBias;
  if (OUT_F) Cf += (long long)batch * sCf;
  if (OUT_B) Cb += (long long)batch * sCb;

  const unsigned ldsA = (unsigned)(uintptr_t)(&As[0][0]);
  const unsigned ldsB = (unsigned)(uintptr_t)(&Bs[0][0]);
  const unsigned BUFB = 128u * 40u * 2u;  // bytes per buffer

  f32x8 acc[2][4];
  #pragma unroll
  for (int i = 0; i < 2; ++i)
    #pragma unroll
    for (int j = 0; j < 4; ++j) acc[i][j] = zero8();

  // Preload buffer 0
  async_tile_128x32(Ab, ldsA, K, tid);
  async_tile_128x32(Bb, ldsB, K, tid);

  const int nsteps = K >> 5;
  for (int s2 = 0; s2 < nsteps; ++s2) {
    const int buf = s2 & 1;
    if (s2 + 1 < nsteps) {
      // Prefetch next tile pair into the other buffer, then retire the
      // oldest 4 async ops (current buffer) while the new 4 stay in flight.
      async_tile_128x32(Ab + (s2 + 1) * 32, ldsA + (unsigned)(buf ^ 1) * BUFB, K, tid);
      async_tile_128x32(Bb + (s2 + 1) * 32, ldsB + (unsigned)(buf ^ 1) * BUFB, K, tid);
      wait_async_le4();
    } else {
      wait_async_0();
    }
    __syncthreads();  // all waves' async writes to current buffer visible

    const __bf16* Asb = &As[buf][0];
    const __bf16* Bsb = &Bs[buf][0];
    FragBF a[2];
    #pragma unroll
    for (int i = 0; i < 2; ++i) {
      const int r = wm + i * 16 + lh;
      a[i].q[0] = *(const uint4*)(&Asb[r * 40 + kh * 8]);
      a[i].q[1] = *(const uint4*)(&Asb[r * 40 + 16 + kh * 8]);
    }
    #pragma unroll
    for (int j = 0; j < 4; ++j) {
      const int n = wn + j * 16 + lh;
      FragBF b;
      b.q[0] = *(const uint4*)(&Bsb[n * 40 + kh * 8]);
      b.q[1] = *(const uint4*)(&Bsb[n * 40 + 16 + kh * 8]);
      #pragma unroll
      for (int i = 0; i < 2; ++i)
        acc[i][j] = __builtin_amdgcn_wmma_f32_16x16x32_bf16(
            false, a[i].v, false, b.v, (short)0, acc[i][j], false, false);
    }
    __syncthreads();  // protect buffer from next-next prefetch overwrite
  }

  // Epilogue (template-specialized: no runtime branches in the loops)
  #pragma unroll
  for (int j = 0; j < 4; ++j) {
    const int n = n0 + wn + j * 16 + lh;
    const float bv = HAS_BIAS ? bias[n] : 0.f;
    #pragma unroll
    for (int i = 0; i < 2; ++i) {
      #pragma unroll
      for (int v = 0; v < 8; ++v) {
        const int m = m0 + wm + i * 16 + v + 8 * kh;
        float val = acc[i][j][v] + bv;
        if (RELU) val = fmaxf(val, 0.f);
        const long long off = (long long)m * N + n;
        if (OUT_F) Cf[off] = val;
        if (OUT_B) Cb[off] = (__bf16)val;
      }
    }
  }
}

// ---------------------------------------------------------------------------
// Split QKV (f32, [b*T+t][3*1024]) into q_bf (pre-scaled by 1/8), k_bf
// ([b,h,t,d]) and vT_bf ([b,h,d,t]) for WMMA-friendly contiguous loads.
// ---------------------------------------------------------------------------
__global__ __launch_bounds__(256)
void k_qkv_split(const float* __restrict__ qkv, __bf16* __restrict__ qb,
                 __bf16* __restrict__ kb, __bf16* __restrict__ vtb) {
  long long i = (long long)blockIdx.x * 256 + threadIdx.x;
  const long long n = 4194304ll;  // 4096*1024
  const long long stride = (long long)gridDim.x * 256;
  for (; i < n; i += stride) {
    const int bt = (int)(i >> 10);
    const int c = (int)(i & 1023);
    const int b = bt >> 11, t = bt & 2047;
    const int h = c >> 6, d = c & 63;
    const float* base = qkv + (long long)bt * 3072;
    const float q = base[c];
    const float kk = base[1024 + c];
    const float vv = base[2048 + c];
    const long long bh = (long long)(b * 16 + h);
    qb[(bh * 2048 + t) * 64 + d] = (__bf16)(q * 0.125f);
    kb[(bh * 2048 + t) * 64 + d] = (__bf16)kk;
    vtb[(bh * 64 + d) * 2048 + t] = (__bf16)vv;
  }
}

// ---------------------------------------------------------------------------
// Attention: one wave per 16 query rows, streaming over 32-key tiles.
// Scores via 2 WMMAs, exp on C regs (q pre-scaled by 1/8, activations O(1),
// so scores are small and max-subtraction is safely skipped in f32), P staged
// through wave-private LDS to re-layout C->A fragment, O += P@V via 4 WMMAs
// against pre-transposed V. Output stored with the reference reshape scramble.
// ---------------------------------------------------------------------------
__global__ __launch_bounds__(256)
void k_attention(const __bf16* __restrict__ qb, const __bf16* __restrict__ kb,
                 const __bf16* __restrict__ vtb, __bf16* __restrict__ attnb) {
  __shared__ __align__(16) __bf16 Pst[8][16 * 40];
  const int b = blockIdx.z, h = blockIdx.y;
  const int tid = threadIdx.x, wave = tid >> 5, lane = tid & 31;
  const int lh = lane & 15, khf = lane >> 4;
  const int t0 = blockIdx.x * 128 + wave * 16;
  const long long bh = (long long)(b * 16 + h);

  const __bf16* qrow = qb + (bh * 2048 + t0 + lh) * 64;
  FragBF a0, a1;
  a0.q[0] = *(const uint4*)(qrow + khf * 8);
  a0.q[1] = *(const uint4*)(qrow + 16 + khf * 8);
  a1.q[0] = *(const uint4*)(qrow + 32 + khf * 8);
  a1.q[1] = *(const uint4*)(qrow + 48 + khf * 8);

  f32x8 O[4];
  #pragma unroll
  for (int j = 0; j < 4; ++j) O[j] = zero8();
  f32x8 sp = zero8();

  const __bf16* kbase = kb + bh * 2048 * 64;
  const __bf16* vbase = vtb + bh * 64 * 2048;
  __bf16* Pw = &Pst[wave][0];

  for (int kt = 0; kt < 2048; kt += 32) {
    #pragma unroll
    for (int sub = 0; sub < 2; ++sub) {
      const __bf16* kr = kbase + (long long)(kt + sub * 16 + lh) * 64 + khf * 8;
      FragBF b0, b1;
      b0.q[0] = *(const uint4*)(kr);
      b0.q[1] = *(const uint4*)(kr + 16);
      b1.q[0] = *(const uint4*)(kr + 32);
      b1.q[1] = *(const uint4*)(kr + 48);
      f32x8 c = zero8();
      c = __builtin_amdgcn_wmma_f32_16x16x32_bf16(false, a0.v, false, b0.v, (short)0, c, false, false);
      c = __builtin_amdgcn_wmma_f32_16x16x32_bf16(false, a1.v, false, b1.v, (short)0, c, false, false);
      #pragma unroll
      for (int v = 0; v < 8; ++v) {
        const float e = __expf(c[v]);
        sp[v] += e;
        Pw[(v + 8 * khf) * 40 + sub * 16 + lh] = (__bf16)e;
      }
    }
    FragBF pf;
    pf.q[0] = *(const uint4*)(&Pw[lh * 40 + khf * 8]);
    pf.q[1] = *(const uint4*)(&Pw[lh * 40 + 16 + khf * 8]);
    #pragma unroll
    for (int j = 0; j < 4; ++j) {
      const __bf16* vr = vbase + (long long)(j * 16 + lh) * 2048 + kt + khf * 8;
      FragBF bv;
      bv.q[0] = *(const uint4*)(vr);
      bv.q[1] = *(const uint4*)(vr + 16);
      O[j] = __builtin_amdgcn_wmma_f32_16x16x32_bf16(false, pf.v, false, bv.v, (short)0, O[j], false, false);
    }
  }

  float inv[8];
  #pragma unroll
  for (int v = 0; v < 8; ++v) {
    float s = sp[v];
    #pragma unroll
    for (int off = 1; off < 16; off <<= 1) s += __shfl_xor(s, off, 16);
    inv[v] = 1.f / s;
  }

  // t' = (t%32)*64 + d ; c' = h*64 + t/32  (reference reshape scramble)
  #pragma unroll
  for (int j = 0; j < 4; ++j) {
    const int d = j * 16 + lh;
    #pragma unroll
    for (int v = 0; v < 8; ++v) {
      const int t = t0 + v + 8 * khf;
      const float val = O[j][v] * inv[v];
      const int tp = (t & 31) * 64 + d;
      const int cp = h * 64 + (t >> 5);
      attnb[((long long)b * 2048 + tp) * 1024 + cp] = (__bf16)val;
    }
  }
}

// ---------------------------------------------------------------------------
// Router: per-token logits/noise-logits (K=1024, E=8), noisy top-2, gates,
// entropy of full softmax. Also initializes slot_map = -1.
// ---------------------------------------------------------------------------
__global__ __launch_bounds__(128)
void k_route(const float* __restrict__ h2, const float* __restrict__ wr,
             const float* __restrict__ br, const float* __restrict__ wn,
             const float* __restrict__ bn, const float* __restrict__ noise,
             float* __restrict__ ent, int* __restrict__ topidx,
             float* __restrict__ topg, int* __restrict__ smap) {
  __shared__ float red[16][128];
  const int t = blockIdx.x, tid = threadIdx.x;
  float pr[8], pn[8];
  #pragma unroll
  for (int e = 0; e < 8; ++e) { pr[e] = 0.f; pn[e] = 0.f; }
  const float* xr = h2 + (long long)t * 1024;
  for (int k = tid; k < 1024; k += 128) {
    const float xv = xr[k];
    const float* wrk = wr + k * 8;
    const float* wnk = wn + k * 8;
    #pragma unroll
    for (int e = 0; e < 8; ++e) {
      pr[e] += xv * wrk[e];
      pn[e] += xv * wnk[e];
    }
  }
  #pragma unroll
  for (int e = 0; e < 8; ++e) {
    red[e][tid] = pr[e];
    red[8 + e][tid] = pn[e];
  }
  __syncthreads();
  if (tid < 16) {
    float s = 0.f;
    for (int j = 0; j < 128; ++j) s += red[tid][j];
    red[tid][0] = s;
  }
  __syncthreads();
  if (tid == 0) {
    float noisy[8], probs[8];
    #pragma unroll
    for (int e = 0; e < 8; ++e) {
      const float lg = red[e][0] + br[e];
      const float nl = red[8 + e][0] + bn[e];
      const float spv = (nl > 20.f) ? nl : log1pf(__expf(nl));  // softplus
      noisy[e] = lg + noise[(long long)t * 8 + e] * spv;
    }
    float mx = noisy[0];
    #pragma unroll
    for (int e = 1; e < 8; ++e) mx = fmaxf(mx, noisy[e]);
    float se = 0.f;
    #pragma unroll
    for (int e = 0; e < 8; ++e) { probs[e] = __expf(noisy[e] - mx); se += probs[e]; }
    const float invs = 1.f / se;
    float H = 0.f;
    #pragma unroll
    for (int e = 0; e < 8; ++e) {
      const float p = probs[e] * invs;
      H -= p * logf(p + 1e-8f);
    }
    ent[t] = H;
    int i0 = 0;
    #pragma unroll
    for (int e = 1; e < 8; ++e) if (noisy[e] > noisy[i0]) i0 = e;
    int i1 = (i0 == 0) ? 1 : 0;
    #pragma unroll
    for (int e = 0; e < 8; ++e)
      if (e != i0 && noisy[e] > noisy[i1]) i1 = e;
    const float g0 = 1.f / (1.f + __expf(noisy[i1] - noisy[i0]));
    topidx[2 * t] = i0;
    topidx[2 * t + 1] = i1;
    topg[2 * t] = g0;
    topg[2 * t + 1] = 1.f - g0;
    smap[2 * t] = -1;
    smap[2 * t + 1] = -1;
  }
}

// ---------------------------------------------------------------------------
// Per-expert stable prefix-scan; first 1024 selectors (token order) get a
// slot (cap = N*TOP_K/E = 1024). Deterministic (no atomics).
// ---------------------------------------------------------------------------
__global__ __launch_bounds__(1024)
void k_scan(const int* __restrict__ topidx, const float* __restrict__ topg,
            int* __restrict__ etok, int* __restrict__ smap) {
  __shared__ int s[1024];
  const int e = blockIdx.x, tid = threadIdx.x;
  etok[e * 1024 + tid] = 0;  // pad slots gather token 0 (never combined back)
  __syncthreads();
  int carry = 0;
  for (int chunk = 0; chunk < 4; ++chunk) {
    const int t = chunk * 1024 + tid;
    const int i0 = topidx[2 * t], i1 = topidx[2 * t + 1];
    const int which = (i0 == e) ? 0 : ((i1 == e) ? 1 : -1);
    const int v = (which >= 0) ? 1 : 0;
    s[tid] = v;
    __syncthreads();
    for (int off = 1; off < 1024; off <<= 1) {
      const int add = (tid >= off) ? s[tid - off] : 0;
      __syncthreads();
      s[tid] += add;
      __syncthreads();
    }
    const int pos = carry + s[tid] - v;  // exclusive prefix
    if (which >= 0 && pos < 1024) {
      etok[e * 1024 + pos] = t;
      smap[2 * t + which] = e * 1024 + pos;
    }
    carry += s[1023];
    __syncthreads();
  }
}

// ---------------------------------------------------------------------------
// Gather selected token rows (bf16) into per-expert input matrices.
// ---------------------------------------------------------------------------
__global__ __launch_bounds__(128)
void k_gather(const __bf16* __restrict__ h2b, const int* __restrict__ etok,
              __bf16* __restrict__ xib) {
  const int row = blockIdx.x;  // expert*1024 + slot
  const int tok = etok[row];
  const uint4* src = (const uint4*)(h2b + (long long)tok * 1024);
  uint4* dst = (uint4*)(xib + (long long)row * 1024);
  dst[threadIdx.x] = src[threadIdx.x];
}

// ---------------------------------------------------------------------------
// Combine: out = x1 + sum_k gate_k * yo[slot_k]  (deterministic gather)
// ---------------------------------------------------------------------------
__global__ __launch_bounds__(256)
void k_combine(const float* __restrict__ x1, const float* __restrict__ yof,
               const int* __restrict__ smap, const float* __restrict__ topg,
               float* __restrict__ out) {
  const int t = blockIdx.x, tid = threadIdx.x;
  const int s0 = smap[2 * t], s1 = smap[2 * t + 1];
  const float g0 = topg[2 * t], g1 = topg[2 * t + 1];
  float4 acc = ((const float4*)(x1 + (long long)t * 1024))[tid];
  if (s0 >= 0) {
    float4 y = ((const float4*)(yof + (long long)s0 * 1024))[tid];
    acc.x += g0 * y.x; acc.y += g0 * y.y; acc.z += g0 * y.z; acc.w += g0 * y.w;
  }
  if (s1 >= 0) {
    float4 y = ((const float4*)(yof + (long long)s1 * 1024))[tid];
    acc.x += g1 * y.x; acc.y += g1 * y.y; acc.z += g1 * y.z; acc.w += g1 * y.w;
  }
  ((float4*)(out + (long long)t * 1024))[tid] = acc;
}

// ---------------------------------------------------------------------------
// Deterministic entropy mean over 4096 tokens.
// ---------------------------------------------------------------------------
__global__ __launch_bounds__(1024)
void k_entropy(const float* __restrict__ ent, float* __restrict__ out_scalar) {
  __shared__ float r[1024];
  const int tid = threadIdx.x;
  r[tid] = ent[tid] + ent[tid + 1024] + ent[tid + 2048] + ent[tid + 3072];
  __syncthreads();
  for (int off = 512; off > 0; off >>= 1) {
    if (tid < off) r[tid] += r[tid + off];
    __syncthreads();
  }
  if (tid == 0) out_scalar[0] = r[0] * (1.f / 4096.f);
}

// ---------------------------------------------------------------------------
// Host orchestration
// ---------------------------------------------------------------------------
extern "C" void kernel_launch(void* const* d_in, const int* in_sizes, int n_in,
                              void* d_out, int out_size, void* d_ws, size_t ws_size,
                              hipStream_t stream) {
  (void)in_sizes; (void)n_in; (void)out_size; (void)ws_size;
  const float* x       = (const float*)d_in[0];
  const float* noise   = (const float*)d_in[1];
  const float* gamma1  = (const float*)d_in[2];
  const float* beta1   = (const float*)d_in[3];
  const float* gamma2  = (const float*)d_in[4];
  const float* beta2   = (const float*)d_in[5];
  const float* w_qkv   = (const float*)d_in[6];
  const float* w_out   = (const float*)d_in[7];
  const float* w_route = (const float*)d_in[8];
  const float* b_route = (const float*)d_in[9];
  const float* w_noise = (const float*)d_in[10];
  const float* b_noise = (const float*)d_in[11];
  const float* w1      = (const float*)d_in[12];
  const float* b1      = (const float*)d_in[13];
  const float* w2      = (const float*)d_in[14];
  const float* b2      = (const float*)d_in[15];
  float* out = (float*)d_out;

  char* ws = (char*)d_ws;
  size_t off = 0;
  auto alloc = [&](size_t bytes) -> char* {
    char* p = ws + off;
    off += (bytes + 255) & ~(size_t)255;
    return p;
  };

  // Transposed bf16 weights (BT layout [N][K] for async-LDS GEMM staging)
  __bf16* wqkvT = (__bf16*)alloc(3072ll * 1024 * 2);
  __bf16* woutT = (__bf16*)alloc(1024ll * 1024 * 2);
  __bf16* w1T   = (__bf16*)alloc(8ll * 4096 * 1024 * 2);
  __bf16* w2T   = (__bf16*)alloc(8ll * 1024 * 4096 * 2);
  __bf16* h1b   = (__bf16*)alloc(4096ll * 1024 * 2);
  float*  qkvf  = (float*)alloc(4096ll * 3072 * 4);
  __bf16* qb    = (__bf16*)alloc(4096ll * 1024 * 2);
  __bf16* kbuf  = (__bf16*)alloc(4096ll * 1024 * 2);
  __bf16* vtb   = (__bf16*)alloc(4096ll * 1024 * 2);
  __bf16* attnb = (__bf16*)alloc(4096ll * 1024 * 2);
  float*  projf = (float*)alloc(4096ll * 1024 * 4);
  float*  x1f   = (float*)alloc(4096ll * 1024 * 4);
  float*  h2f   = (float*)alloc(4096ll * 1024 * 4);
  __bf16* h2b   = (__bf16*)alloc(4096ll * 1024 * 2);
  __bf16* xib   = (__bf16*)alloc(8ll * 1024 * 1024 * 2);
  __bf16* hidb  = (__bf16*)alloc(8ll * 1024 * 4096 * 2);
  float*  yof   = (float*)alloc(8ll * 1024 * 1024 * 4);
  float*  entw  = (float*)alloc(4096 * 4);
  int*    topidx= (int*)alloc(4096 * 2 * 4);
  float*  topg  = (float*)alloc(4096 * 2 * 4);
  int*    etok  = (int*)alloc(8192 * 4);
  int*    smap  = (int*)alloc(8192 * 4);

  // Weight conversion f32 -> bf16 with transpose: dst[N][K]
  k_cvt_t<<<dim3(96, 32, 1),  256, 0, stream>>>(w_qkv, wqkvT, 1024, 3072);
  k_cvt_t<<<dim3(32, 32, 1),  256, 0, stream>>>(w_out, woutT, 1024, 1024);
  k_cvt_t<<<dim3(128, 32, 8), 256, 0, stream>>>(w1, w1T, 1024, 4096);
  k_cvt_t<<<dim3(32, 128, 8), 256, 0, stream>>>(w2, w2T, 4096, 1024);

  // LN1
  k_ln_bf16<<<4096, 256, 0, stream>>>(x, nullptr, gamma1, beta1,
                                      nullptr, nullptr, h1b);
  // QKV GEMM: [4096,1024] x [1024,3072]
  k_gemm_bf16<false, false, true, false><<<dim3(24, 32, 1), 256, 0, stream>>>(
      h1b, wqkvT, nullptr, qkvf, nullptr, 4096, 3072, 1024, 0, 0, 0, 0, 0);
  // Split + layouts for attention
  k_qkv_split<<<8192, 256, 0, stream>>>(qkvf, qb, kbuf, vtb);
  // Attention
  k_attention<<<dim3(16, 16, 2), 256, 0, stream>>>(qb, kbuf, vtb, attnb);
  // Output projection: [4096,1024] x [1024,1024]
  k_gemm_bf16<false, false, true, false><<<dim3(8, 32, 1), 256, 0, stream>>>(
      attnb, woutT, nullptr, projf, nullptr, 4096, 1024, 1024, 0, 0, 0, 0, 0);
  // Residual + LN2
  k_ln_bf16<<<4096, 256, 0, stream>>>(x, projf, gamma2, beta2, x1f, h2f, h2b);
  // Routing
  k_route<<<4096, 128, 0, stream>>>(h2f, w_route, b_route, w_noise, b_noise,
                                    noise, entw, topidx, topg, smap);
  // Capacity-limited expert assignment (deterministic scan)
  k_scan<<<8, 1024, 0, stream>>>(topidx, topg, etok, smap);
  // Gather expert inputs
  k_gather<<<8192, 128, 0, stream>>>(h2b, etok, xib);
  // Expert FFN layer 1 (batched, bias+relu, bf16 out): [1024,1024]x[1024,4096]
  k_gemm_bf16<true, true, false, true><<<dim3(32, 8, 8), 256, 0, stream>>>(
      xib, w1T, b1, nullptr, hidb, 1024, 4096, 1024,
      1024ll * 1024, 4096ll * 1024, 4096, 0, 1024ll * 4096);
  // Expert FFN layer 2 (batched, bias, f32 out): [1024,4096]x[4096,1024]
  k_gemm_bf16<true, false, true, false><<<dim3(8, 8, 8), 256, 0, stream>>>(
      hidb, w2T, b2, yof, nullptr, 1024, 1024, 4096,
      1024ll * 4096, 1024ll * 4096, 1024, 1024ll * 1024, 0);
  // Gated combine + residual
  k_combine<<<4096, 256, 0, stream>>>(x1f, yof, smap, topg, out);
  // Entropy mean scalar
  k_entropy<<<1, 1024, 0, stream>>>(entw, out + 4194304ll);
}
